// AttentionBlock_67508295959402
// MI455X (gfx1250) — compile-verified, matrix operand order checked by reading, not run on hardware
//
#include <hip/hip_runtime.h>
#include <stdint.h>

// ---------------------------------------------------------------------------
// AttentionBlock for MI455X (gfx1250): LN -> QKV gemm -> flash attn -> out gemm
// All matmuls use v_wmma_f32_16x16x32_bf16 (wave32, fp32 accumulate).
// Double-buffered LDS staging; async global->LDS copies when available.
// B=4 S=2048 D=1024 H=16 HD=64 (hardcoded per reference).
// ---------------------------------------------------------------------------

typedef __attribute__((ext_vector_type(16))) __bf16 v16bf;
typedef __attribute__((ext_vector_type(8)))  float  v8f;

union FragBF { v16bf v; uint32_t u[8]; };
union U4U16  { uint4 q; unsigned short s[8]; };

#ifndef __has_builtin
#define __has_builtin(x) 0
#endif
#if __has_builtin(__builtin_amdgcn_global_load_async_to_lds_b128) && \
    __has_builtin(__builtin_amdgcn_s_wait_asynccnt)
#define ASYNC_LDS 1
#else
#define ASYNC_LDS 0
#endif

#if ASYNC_LDS
// 16-byte global -> LDS async copy (tracked by ASYNCcnt, no VGPR round-trip).
// Builtin signature (from hipcc diagnostic): (int4 AS1*, int4 AS3*, imm, imm).
// Generic->AS casts via integer (AS1 numerically identical; AS3 = low 32 bits).
typedef int v4i_t __attribute__((vector_size(16)));
typedef v4i_t __attribute__((address_space(1))) glb_v4i;
typedef v4i_t __attribute__((address_space(3))) lds_v4i;
__device__ __forceinline__ void async_copy16(const void* g, void* l) {
  __builtin_amdgcn_global_load_async_to_lds_b128(
      (glb_v4i*)(uintptr_t)g,
      (lds_v4i*)(uint32_t)(uintptr_t)l, 0, 0);
}
__device__ __forceinline__ void async_wait() { __builtin_amdgcn_s_wait_asynccnt(0); }
#else
__device__ __forceinline__ void async_copy16(const void* g, void* l) {
  *reinterpret_cast<uint4*>(l) = *reinterpret_cast<const uint4*>(g);
}
__device__ __forceinline__ void async_wait() {}
#endif

__device__ __forceinline__ unsigned short f2bf(float f) {
  union { float f; uint32_t u; } c; c.f = f;
  uint32_t r = c.u + 0x7fffu + ((c.u >> 16) & 1u);   // round-to-nearest-even
  return (unsigned short)(r >> 16);
}

// ISA 16-bit A/B fragment K-swizzle: VGPR p, lane-half h -> K offset in [0,32)
__device__ __forceinline__ int kswz(int p, int hf) {
  return ((p & 4) << 2) + hf * 8 + ((p & 3) << 1);
}

// ---------------------------------------------------------------------------
// 1) fp32 -> bf16 cast (weights), 4 elems / thread
// ---------------------------------------------------------------------------
__global__ __launch_bounds__(256) void cast_bf16_kernel(
    const float* __restrict__ src, unsigned short* __restrict__ dst) {
  const size_t i = (size_t)blockIdx.x * 256 + threadIdx.x;
  const float4 d = reinterpret_cast<const float4*>(src)[i];
  ushort4 o;
  o.x = f2bf(d.x); o.y = f2bf(d.y); o.z = f2bf(d.z); o.w = f2bf(d.w);
  reinterpret_cast<ushort4*>(dst)[i] = o;
}

// ---------------------------------------------------------------------------
// 2) LayerNorm over D=1024 (one row / block, 256 threads x 4 elems) -> bf16
// ---------------------------------------------------------------------------
__global__ __launch_bounds__(256) void ln_cast_kernel(
    const float* __restrict__ x, const float* __restrict__ sc,
    const float* __restrict__ bi, unsigned short* __restrict__ xn) {
  const int row = blockIdx.x;
  const int tid = threadIdx.x;
  const float4 d = reinterpret_cast<const float4*>(x + (size_t)row * 1024)[tid];
  float s  = d.x + d.y + d.z + d.w;
  float s2 = d.x * d.x + d.y * d.y + d.z * d.z + d.w * d.w;
  #pragma unroll
  for (int m = 16; m >= 1; m >>= 1) {
    s  += __shfl_xor(s,  m, 32);
    s2 += __shfl_xor(s2, m, 32);
  }
  __shared__ float red[2][8];
  const int wave = tid >> 5, lane = tid & 31;
  if (lane == 0) { red[0][wave] = s; red[1][wave] = s2; }
  __syncthreads();
  float ts = 0.f, ts2 = 0.f;
  #pragma unroll
  for (int w = 0; w < 8; ++w) { ts += red[0][w]; ts2 += red[1][w]; }
  const float mu  = ts * (1.0f / 1024.0f);
  const float var = ts2 * (1.0f / 1024.0f) - mu * mu;
  const float rs  = rsqrtf(var + 1e-6f);
  const int n0 = tid * 4;
  ushort4 o;
  o.x = f2bf((d.x - mu) * rs * sc[n0 + 0] + bi[n0 + 0]);
  o.y = f2bf((d.y - mu) * rs * sc[n0 + 1] + bi[n0 + 1]);
  o.z = f2bf((d.z - mu) * rs * sc[n0 + 2] + bi[n0 + 2]);
  o.w = f2bf((d.w - mu) * rs * sc[n0 + 3] + bi[n0 + 3]);
  reinterpret_cast<ushort4*>(xn + (size_t)row * 1024)[tid] = o;
}

// ---------------------------------------------------------------------------
// 3/5) bf16 GEMM  C[M=8192, N] = A[M,1024] x B[1024,N] + bias
//     block tile 128x128, 8 waves, wave tile 64x32 (4x2 WMMA tiles), K-step 32
//     double-buffered LDS, async A-stage, prefetch 2 steps ahead
//     MODE 0: QKV epilogue (scale q, scatter q/k/v bf16 as [B,H,S,64])
//     MODE 1: fp32 store to out
// ---------------------------------------------------------------------------
template <int MODE>
__global__ __launch_bounds__(256) void gemm128_bf16(
    const unsigned short* __restrict__ A,
    const unsigned short* __restrict__ Bw,
    const float* __restrict__ bias,
    unsigned short* __restrict__ qd, unsigned short* __restrict__ kd,
    unsigned short* __restrict__ vd, float* __restrict__ outF, const int N) {
  __shared__ unsigned short lA[2][128 * 40];   // [m][k], stride 40 (16B aligned)
  __shared__ unsigned short lB[2][128 * 40];   // [n][k] transposed, stride 40
  const int tid  = threadIdx.x;
  const int wave = tid >> 5, lane = tid & 31;
  const int hf = lane >> 4, ln = lane & 15;
  const int wm = wave >> 2, wn = wave & 3;
  const int bm = blockIdx.y, bn = blockIdx.x;

  const v8f vz = {0.f, 0.f, 0.f, 0.f, 0.f, 0.f, 0.f, 0.f};
  v8f acc[4][2];
  #pragma unroll
  for (int i = 0; i < 4; ++i)
    #pragma unroll
    for (int j = 0; j < 2; ++j) acc[i][j] = vz;

  // tile stagers (A contiguous -> async capable; B transposed -> manual)
  auto stageA = [&](int k0, int buf) {
    #pragma unroll
    for (int rr = 0; rr < 2; ++rr) {
      const int idx = tid + rr * 256;
      const int ar = idx >> 2, ac = (idx & 3) * 8;
      async_copy16(A + (size_t)(bm * 128 + ar) * 1024 + k0 + ac,
                   &lA[buf][ar * 40 + ac]);
    }
  };
  auto stageB = [&](int k0, int buf) {
    #pragma unroll
    for (int rr = 0; rr < 2; ++rr) {
      const int idx = tid + rr * 256;
      const int kk = idx >> 4, nc = (idx & 15) * 8;
      U4U16 dv;
      dv.q = *reinterpret_cast<const uint4*>(
          Bw + (size_t)(k0 + kk) * N + bn * 128 + nc);
      #pragma unroll
      for (int e = 0; e < 8; ++e) lB[buf][(nc + e) * 40 + kk] = dv.s[e];
    }
  };

  stageA(0, 0);
  stageB(0, 0);
  async_wait();
  __syncthreads();

  for (int k0 = 0; k0 < 1024; k0 += 32) {
    const int cur = (k0 >> 5) & 1;
    if (k0 + 32 < 1024) {           // stage next tile into the other buffer
      stageA(k0 + 32, cur ^ 1);
      stageB(k0 + 32, cur ^ 1);
    }
    if (k0 + 64 < 1024) {           // warm GL2 two steps ahead
      __builtin_prefetch(A + (size_t)(bm * 128 + (tid >> 1)) * 1024 + k0 + 64, 0, 1);
      __builtin_prefetch(Bw + (size_t)(k0 + 64 + (tid & 31)) * N + bn * 128, 0, 1);
    }

    FragBF fa[4], fb[2];
    #pragma unroll
    for (int ti = 0; ti < 4; ++ti) {
      const int row = wm * 64 + ti * 16 + ln;
      #pragma unroll
      for (int p = 0; p < 8; ++p)
        fa[ti].u[p] = *reinterpret_cast<const uint32_t*>(&lA[cur][row * 40 + kswz(p, hf)]);
    }
    #pragma unroll
    for (int tj = 0; tj < 2; ++tj) {
      const int nn = wn * 32 + tj * 16 + ln;
      #pragma unroll
      for (int p = 0; p < 8; ++p)
        fb[tj].u[p] = *reinterpret_cast<const uint32_t*>(&lB[cur][nn * 40 + kswz(p, hf)]);
    }
    #pragma unroll
    for (int ti = 0; ti < 4; ++ti)
      #pragma unroll
      for (int tj = 0; tj < 2; ++tj)
        acc[ti][tj] = __builtin_amdgcn_wmma_f32_16x16x32_bf16(
            false, fa[ti].v, false, fb[tj].v, (short)0, acc[ti][tj], false, false);

    async_wait();
    __syncthreads();
  }

  // epilogue (C layout: M = r + 8*half, N = lane&15 within each 16x16 tile)
  #pragma unroll
  for (int ti = 0; ti < 4; ++ti)
    #pragma unroll
    for (int tj = 0; tj < 2; ++tj)
      #pragma unroll
      for (int r = 0; r < 8; ++r) {
        const int m = bm * 128 + wm * 64 + ti * 16 + r + hf * 8;
        const int n = bn * 128 + wn * 32 + tj * 16 + ln;
        float val = acc[ti][tj][r] + bias[n];
        if (MODE == 0) {
          const int h = n / 192, r2 = n - h * 192;
          const int which = r2 >> 6, hd = r2 & 63;
          if (which == 0) val *= 0.125f;  // 1/sqrt(64)
          const int b = m >> 11, srow = m & 2047;
          const size_t di = (((size_t)b * 16 + h) * 2048 + srow) * 64 + hd;
          const unsigned short bv = f2bf(val);
          if (which == 0) qd[di] = bv;
          else if (which == 1) kd[di] = bv;
          else vd[di] = bv;
        } else {
          outF[(size_t)m * N + n] = val;
        }
      }
}

// ---------------------------------------------------------------------------
// 4) Flash attention: one (128-row q block) x (b,h) per block; 8 waves x 16 rows
//    KV blocks of 64 keys, double-buffered; WMMA for Q*K^T and P*V.
// ---------------------------------------------------------------------------
__global__ __launch_bounds__(256) void attn_kernel(
    const unsigned short* __restrict__ q, const unsigned short* __restrict__ k,
    const unsigned short* __restrict__ v, unsigned short* __restrict__ y) {
  __shared__ unsigned short lK[2][64 * 64];     // [key][hd]
  __shared__ unsigned short lV[2][64 * 72];     // [hd][key] transposed, pad 72
  __shared__ unsigned short lP[8][16 * 64];     // per-wave P tile [row][key]
  const int tid  = threadIdx.x;
  const int wave = tid >> 5, lane = tid & 31;
  const int hf = lane >> 4, ln = lane & 15;
  const int qb = blockIdx.x, bh = blockIdx.y;
  const int b = bh >> 4, h = bh & 15;
  const size_t base = (size_t)bh * 2048 * 64;
  const int qbase = qb * 128 + wave * 16;

  // Q fragments for this wave's 16 rows (A layout: M = lane&15)
  FragBF aq[2];
  #pragma unroll
  for (int kq = 0; kq < 2; ++kq)
    #pragma unroll
    for (int p = 0; p < 8; ++p) {
      const int hd = kq * 32 + kswz(p, hf);
      aq[kq].u[p] = *reinterpret_cast<const uint32_t*>(
          q + base + (size_t)(qbase + ln) * 64 + hd);
    }

  const v8f vz = {0.f, 0.f, 0.f, 0.f, 0.f, 0.f, 0.f, 0.f};
  float m_i[8], l_i[8];
  v8f o[4];
  #pragma unroll
  for (int r = 0; r < 8; ++r) { m_i[r] = -3.0e38f; l_i[r] = 0.f; }
  #pragma unroll
  for (int t = 0; t < 4; ++t) o[t] = vz;

  auto stageKV = [&](int jb, int buf) {
    const int kvbase = jb * 64;
    #pragma unroll
    for (int rr = 0; rr < 2; ++rr) {
      const int idx = tid + rr * 256;
      const int key = idx >> 3, cg = (idx & 7) * 8;
      async_copy16(k + base + (size_t)(kvbase + key) * 64 + cg,
                   &lK[buf][key * 64 + cg]);                     // K direct
      U4U16 dv;                                                  // V transposed
      dv.q = *reinterpret_cast<const uint4*>(
          v + base + (size_t)(kvbase + key) * 64 + cg);
      #pragma unroll
      for (int e = 0; e < 8; ++e) lV[buf][(cg + e) * 72 + key] = dv.s[e];
    }
  };

  const int jbmax = qb * 2 + 1;
  stageKV(0, 0);
  async_wait();
  __syncthreads();

  for (int jb = 0; jb <= jbmax; ++jb) {
    const int kvbase = jb * 64;
    const int cur = jb & 1;
    if (jb < jbmax) stageKV(jb + 1, cur ^ 1);   // overlap next KV with compute

    if (kvbase <= qbase + 15) {   // wave-uniform causal skip (EXEC stays full)
      // scores S[16 x 64] = Q x K^T   (B operand: n=key, K=hd)
      v8f sc[4];
      #pragma unroll
      for (int t = 0; t < 4; ++t) sc[t] = vz;
      #pragma unroll
      for (int t = 0; t < 4; ++t)
        #pragma unroll
        for (int kq = 0; kq < 2; ++kq) {
          FragBF fbk;
          #pragma unroll
          for (int p = 0; p < 8; ++p) {
            const int kk = kq * 32 + kswz(p, hf);
            fbk.u[p] = *reinterpret_cast<const uint32_t*>(
                &lK[cur][(t * 16 + ln) * 64 + kk]);
          }
          sc[t] = __builtin_amdgcn_wmma_f32_16x16x32_bf16(
              false, aq[kq].v, false, fbk.v, (short)0, sc[t], false, false);
        }
      // row max (row = r + 8*hf); mask only on diagonal blocks
      float rowm[8];
      #pragma unroll
      for (int r = 0; r < 8; ++r) rowm[r] = -3.0e38f;
      if (kvbase + 63 <= qbase) {                // fully un-masked block
        #pragma unroll
        for (int t = 0; t < 4; ++t)
          #pragma unroll
          for (int r = 0; r < 8; ++r) rowm[r] = fmaxf(rowm[r], sc[t][r]);
      } else {                                   // diagonal block: causal mask
        #pragma unroll
        for (int t = 0; t < 4; ++t)
          #pragma unroll
          for (int r = 0; r < 8; ++r) {
            const int j = kvbase + t * 16 + ln;
            const int i = qbase + r + hf * 8;
            float sv = (j > i) ? -3.0e38f : sc[t][r];
            sc[t][r] = sv;
            rowm[r] = fmaxf(rowm[r], sv);
          }
      }
      #pragma unroll
      for (int r = 0; r < 8; ++r)
        #pragma unroll
        for (int mk = 1; mk < 16; mk <<= 1)
          rowm[r] = fmaxf(rowm[r], __shfl_xor(rowm[r], mk, 32));

      float alpha[8], rsum[8];
      #pragma unroll
      for (int r = 0; r < 8; ++r) {
        const float mn = fmaxf(m_i[r], rowm[r]);
        alpha[r] = __expf(m_i[r] - mn);
        m_i[r] = mn;
        rsum[r] = 0.f;
      }
      // P = exp(S - m); stash bf16 in per-wave LDS for A-fragment reload
      #pragma unroll
      for (int t = 0; t < 4; ++t)
        #pragma unroll
        for (int r = 0; r < 8; ++r) {
          const float pv = __expf(sc[t][r] - m_i[r]);
          rsum[r] += pv;
          lP[wave][(r + hf * 8) * 64 + t * 16 + ln] = f2bf(pv);
        }
      #pragma unroll
      for (int r = 0; r < 8; ++r) {
        #pragma unroll
        for (int mk = 1; mk < 16; mk <<= 1) rsum[r] += __shfl_xor(rsum[r], mk, 32);
        l_i[r] = l_i[r] * alpha[r] + rsum[r];
      }
      #pragma unroll
      for (int t = 0; t < 4; ++t)
        #pragma unroll
        for (int r = 0; r < 8; ++r) o[t][r] *= alpha[r];

      // O += P x V   (A operand: P rows; B operand: n=hd, K=key from lV)
      FragBF pf[2];
      #pragma unroll
      for (int kq = 0; kq < 2; ++kq)
        #pragma unroll
        for (int p = 0; p < 8; ++p) {
          const int kk = kq * 32 + kswz(p, hf);
          pf[kq].u[p] = *reinterpret_cast<const uint32_t*>(&lP[wave][ln * 64 + kk]);
        }
      #pragma unroll
      for (int t = 0; t < 4; ++t)
        #pragma unroll
        for (int kq = 0; kq < 2; ++kq) {
          FragBF fv;
          #pragma unroll
          for (int p = 0; p < 8; ++p) {
            const int kk = kq * 32 + kswz(p, hf);
            fv.u[p] = *reinterpret_cast<const uint32_t*>(
                &lV[cur][(t * 16 + ln) * 72 + kk]);
          }
          o[t] = __builtin_amdgcn_wmma_f32_16x16x32_bf16(
              false, pf[kq].v, false, fv.v, (short)0, o[t], false, false);
        }
    }
    async_wait();
    __syncthreads();
  }

  // y[b, s, h*64+hd] = O / l   (bf16, feeds out-projection GEMM)
  #pragma unroll
  for (int t = 0; t < 4; ++t)
    #pragma unroll
    for (int r = 0; r < 8; ++r) {
      const float val = o[t][r] / l_i[r];
      const int srow = qbase + r + hf * 8;
      const int hd = t * 16 + ln;
      y[((size_t)(b * 2048 + srow)) * 1024 + h * 64 + hd] = f2bf(val);
    }
}

// ---------------------------------------------------------------------------
// launcher
// ---------------------------------------------------------------------------
extern "C" void kernel_launch(void* const* d_in, const int* in_sizes, int n_in,
                              void* d_out, int out_size, void* d_ws, size_t ws_size,
                              hipStream_t stream) {
  (void)in_sizes; (void)n_in; (void)out_size; (void)ws_size;
  const float* x     = (const float*)d_in[0];
  // d_in[1] = mask (causal, hardcoded)
  const float* ln_s  = (const float*)d_in[2];
  const float* ln_b  = (const float*)d_in[3];
  const float* w_qkv = (const float*)d_in[4];
  const float* b_qkv = (const float*)d_in[5];
  const float* w_out = (const float*)d_in[6];
  const float* b_out = (const float*)d_in[7];
  float* out = (float*)d_out;

  char* ws = (char*)d_ws;
  size_t off = 0;
  auto alloc = [&](size_t bytes) -> void* {
    void* p = ws + off;
    off = (off + bytes + 255) & ~(size_t)255;
    return p;
  };
  unsigned short* xn   = (unsigned short*)alloc(8192ull * 1024 * 2);  // LN(x) bf16
  unsigned short* wqkv = (unsigned short*)alloc(1024ull * 3072 * 2);
  unsigned short* wout = (unsigned short*)alloc(1024ull * 1024 * 2);
  unsigned short* qb   = (unsigned short*)alloc(64ull * 2048 * 64 * 2);  // [B*H,S,64]
  unsigned short* kb   = (unsigned short*)alloc(64ull * 2048 * 64 * 2);
  unsigned short* vb   = (unsigned short*)alloc(64ull * 2048 * 64 * 2);
  unsigned short* yb   = (unsigned short*)alloc(8192ull * 1024 * 2);     // attn out bf16

  cast_bf16_kernel<<<3072, 256, 0, stream>>>(w_qkv, wqkv);   // 1024*3072
  cast_bf16_kernel<<<1024, 256, 0, stream>>>(w_out, wout);   // 1024*1024
  ln_cast_kernel<<<8192, 256, 0, stream>>>(x, ln_s, ln_b, xn);
  gemm128_bf16<0><<<dim3(24, 64), 256, 0, stream>>>(xn, wqkv, b_qkv,
                                                    qb, kb, vb, nullptr, 3072);
  attn_kernel<<<dim3(16, 64), 256, 0, stream>>>(qb, kb, vb, yb);
  gemm128_bf16<1><<<dim3(8, 64), 256, 0, stream>>>(yb, wout, b_out,
                                                   nullptr, nullptr, nullptr, out, 1024);
}
